// LFB_NL_18751827214396
// MI455X (gfx1250) — compile-verified, matrix operand order checked by reading, not run on hardware
//
#include <hip/hip_runtime.h>
#include <hip/hip_bf16.h>

// ---------------------------------------------------------------------------
// LFB non-local block for MI455X (gfx1250, wave32).
// Big GEMMs via V_WMMA_F32_16X16X32_BF16 (bf16 inputs, f32 accumulate).
// 64x64 wave tile: 16 WMMA per K-step of 32 from 8 b128 fragment loads.
// ---------------------------------------------------------------------------

typedef __bf16 bf16;
typedef bf16  v16bf __attribute__((ext_vector_type(16)));
typedef float v8f   __attribute__((ext_vector_type(8)));

#define HH 2048
#define NN 512
#define BB 64

// ---- bf16 <-> f32 helpers (bit-level, no reliance on __bf16 arithmetic) ----
__device__ __forceinline__ float bf2f(bf16 x) {
    unsigned short h; __builtin_memcpy(&h, &x, 2);
    unsigned u = ((unsigned)h) << 16;
    float f; __builtin_memcpy(&f, &u, 4);
    return f;
}
__device__ __forceinline__ bf16 f2bf(float f) {
    unsigned u; __builtin_memcpy(&u, &f, 4);
    unsigned r = u + 0x7FFFu + ((u >> 16) & 1u);   // round-to-nearest-even
    unsigned short h = (unsigned short)(r >> 16);
    bf16 o; __builtin_memcpy(&o, &h, 2);
    return o;
}

// ---- f32 -> bf16 conversion kernel ----------------------------------------
__global__ void cvt_f32_bf16(const float* __restrict__ s, bf16* __restrict__ d, int n) {
    int i = blockIdx.x * blockDim.x + threadIdx.x;
    if (i < n) d[i] = f2bf(s[i]);
}

// ---- WMMA fragment load ----------------------------------------------------
// 16-bit A-operand layout (ISA 7.12.2): for the row held by this lane,
//   lanes 0-15 : K = kb+0..7   and K = kb+16..23
//   lanes 16-31: K = kb+8..15  and K = kb+24..31
// Each 8-element chunk is one 16-byte global_load_b128.
union FragU { uint4 u[2]; v16bf v; };

__device__ __forceinline__ v16bf load_frag(const bf16* __restrict__ rowBase, int kb, int half) {
    FragU f;
    const uint4* p = reinterpret_cast<const uint4*>(rowBase + kb + half * 8);
    f.u[0] = p[0];   // K chunk 0
    f.u[1] = p[2];   // K chunk 1 (+16 elements = +32 bytes)
    return f.v;
}

// ---- Generic GEMM:  out[M,N] = X[M,K] @ W[N,K]^T + bias[N] -----------------
// Each wave computes a 64(M) x 64(N) tile: 4x4 accumulators of 16x16,
// 16 v_wmma_f32_16x16x32_bf16 per K-step of 32 from 8 fragment loads.
// outF (f32, optional, addRes accumulates), outB (bf16, optional).
__global__ __launch_bounds__(256)
void gemm_bf16_wmma(const bf16* __restrict__ X, const bf16* __restrict__ W,
                    const float* __restrict__ bias,
                    float* __restrict__ outF, bf16* __restrict__ outB,
                    int M, int N, int K, int addRes) {
    const int lane = threadIdx.x & 31;
    const int half = lane >> 4;     // 0 or 1
    const int lm   = lane & 15;
    const int gw   = blockIdx.x * (blockDim.x >> 5) + (threadIdx.x >> 5);
    const int mTiles = M >> 6;      // 64-row tiles
    const int nTiles = N >> 6;      // 64-col tiles
    if (gw >= mTiles * nTiles) return;
    // waves in a block share the same m-tile over consecutive n-tiles -> A rows
    // get L1/L2 reuse across the 8 waves of the block.
    const int m0 = (gw / nTiles) << 6;
    const int n0 = (gw % nTiles) << 6;

    v8f acc[4][4] = {};

    const bf16* xr[4];
    const bf16* wr[4];
    #pragma unroll
    for (int i = 0; i < 4; ++i) xr[i] = X + (size_t)(m0 + i * 16 + lm) * K;
    #pragma unroll
    for (int j = 0; j < 4; ++j) wr[j] = W + (size_t)(n0 + j * 16 + lm) * K;

    for (int kb = 0; kb < K; kb += 32) {
        v16bf a[4], b[4];
        #pragma unroll
        for (int i = 0; i < 4; ++i) a[i] = load_frag(xr[i], kb, half);
        #pragma unroll
        for (int j = 0; j < 4; ++j) b[j] = load_frag(wr[j], kb, half);
        #pragma unroll
        for (int i = 0; i < 4; ++i) {
            #pragma unroll
            for (int j = 0; j < 4; ++j) {
                acc[i][j] = __builtin_amdgcn_wmma_f32_16x16x32_bf16(
                    false, a[i], false, b[j], (short)0, acc[i][j], false, false);
            }
        }
    }

    // C/D layout: VGPR r holds (M = r + half*8, N = lm) within each 16x16 tile.
    #pragma unroll
    for (int i = 0; i < 4; ++i) {
        #pragma unroll
        for (int j = 0; j < 4; ++j) {
            #pragma unroll
            for (int r = 0; r < 8; ++r) {
                int row = m0 + i * 16 + half * 8 + r;
                int col = n0 + j * 16 + lm;
                float v = acc[i][j][r] + bias[col];
                size_t idx = (size_t)row * N + col;
                if (outF) outF[idx] = addRes ? (outF[idx] + v) : v;
                if (outB) outB[idx] = f2bf(v);
            }
        }
    }
}

// ---- scores[b,n] = scale * sum_h theta[b,h] * phi[b,n,h]  (wave per (b,n)) --
__global__ void scores_kernel(const float* __restrict__ theta, const bf16* __restrict__ phi,
                              float* __restrict__ sc, float scale) {
    int gw = (blockIdx.x * blockDim.x + threadIdx.x) >> 5;
    int lane = threadIdx.x & 31;
    if (gw >= BB * NN) return;
    int b = gw / NN, n = gw % NN;
    const float* th = theta + (size_t)b * HH;
    const bf16* ph = phi + ((size_t)b * NN + n) * HH;
    float s = 0.f;
    for (int h = lane; h < HH; h += 32) s += th[h] * bf2f(ph[h]);
    #pragma unroll
    for (int o = 16; o > 0; o >>= 1) s += __shfl_xor(s, o, 32);
    if (lane == 0) sc[gw] = s * scale;
}

// ---- in-place softmax over rows of 512 (block per row, 256 threads) --------
__global__ __launch_bounds__(256)
void softmax512(float* __restrict__ sc) {
    __shared__ float red[8];
    float* row = sc + (size_t)blockIdx.x * NN;
    int t = threadIdx.x;
    float a = row[t], b = row[t + 256];
    float m = fmaxf(a, b);
    #pragma unroll
    for (int o = 16; o > 0; o >>= 1) m = fmaxf(m, __shfl_xor(m, o, 32));
    if ((t & 31) == 0) red[t >> 5] = m;
    __syncthreads();
    float mx = red[0];
    #pragma unroll
    for (int i = 1; i < 8; ++i) mx = fmaxf(mx, red[i]);
    float e0 = __expf(a - mx), e1 = __expf(b - mx);
    float s = e0 + e1;
    #pragma unroll
    for (int o = 16; o > 0; o >>= 1) s += __shfl_xor(s, o, 32);
    __syncthreads();
    if ((t & 31) == 0) red[t >> 5] = s;
    __syncthreads();
    float tot = 0.f;
    #pragma unroll
    for (int i = 0; i < 8; ++i) tot += red[i];
    float inv = 1.0f / tot;
    row[t] = e0 * inv;
    row[t + 256] = e1 * inv;
}

// ---- t[b,h] = sum_n p[b,n] * g[b,n,h]  (thread per (b,h)) ------------------
__global__ void attend_kernel(const bf16* __restrict__ g, const float* __restrict__ p,
                              float* __restrict__ t) {
    int idx = blockIdx.x * blockDim.x + threadIdx.x;
    if (idx >= BB * HH) return;
    int b = idx / HH, h = idx % HH;
    const bf16* gb = g + (size_t)b * NN * HH + h;
    const float* pb = p + (size_t)b * NN;
    float acc = 0.f;
    for (int n = 0; n < NN; ++n) acc += pb[n] * bf2f(gb[(size_t)n * HH]);
    t[idx] = acc;
}

// ---- layernorm over H (no affine) + ReLU -> bf16 (block per row) -----------
__global__ __launch_bounds__(256)
void ln_relu_kernel(const float* __restrict__ t, bf16* __restrict__ out) {
    __shared__ float rs[8], rss[8];
    const float* x = t + (size_t)blockIdx.x * HH;
    bf16* op = out + (size_t)blockIdx.x * HH;
    int tid = threadIdx.x;
    float v[8], s = 0.f, ss = 0.f;
    #pragma unroll
    for (int i = 0; i < 8; ++i) {
        float q = x[tid + i * 256];
        v[i] = q; s += q; ss += q * q;
    }
    #pragma unroll
    for (int o = 16; o > 0; o >>= 1) { s += __shfl_xor(s, o, 32); ss += __shfl_xor(ss, o, 32); }
    if ((tid & 31) == 0) { rs[tid >> 5] = s; rss[tid >> 5] = ss; }
    __syncthreads();
    float S = 0.f, SS = 0.f;
    #pragma unroll
    for (int i = 0; i < 8; ++i) { S += rs[i]; SS += rss[i]; }
    float mu  = S * (1.0f / (float)HH);
    float var = SS * (1.0f / (float)HH) - mu * mu;
    float inv = rsqrtf(var + 1e-5f);
    #pragma unroll
    for (int i = 0; i < 8; ++i) {
        float q = (v[i] - mu) * inv;
        op[tid + i * 256] = f2bf(fmaxf(q, 0.f));
    }
}

// ---- final: d_out = [A flat (B*H f32), loss=0] -----------------------------
__global__ void finalize_kernel(const float* __restrict__ A, float* __restrict__ out, int n) {
    int i = blockIdx.x * blockDim.x + threadIdx.x;
    if (i < n) out[i] = A[i];
    if (i == 0) out[n] = 0.0f;
}

// ---------------------------------------------------------------------------
extern "C" void kernel_launch(void* const* d_in, const int* in_sizes, int n_in,
                              void* d_out, int out_size, void* d_ws, size_t ws_size,
                              hipStream_t stream) {
    (void)in_sizes; (void)n_in; (void)out_size; (void)ws_size;
    const float* clip = (const float*)d_in[0];
    const float* lfb  = (const float*)d_in[1];
    const float* Wc = (const float*)d_in[2];  const float* bc = (const float*)d_in[3];
    const float* Wl = (const float*)d_in[4];  const float* bl = (const float*)d_in[5];
    const float* Wt = (const float*)d_in[6];  const float* bt = (const float*)d_in[7];
    const float* Wp = (const float*)d_in[8];  const float* bp = (const float*)d_in[9];
    const float* Wg = (const float*)d_in[10]; const float* bg = (const float*)d_in[11];
    const float* Wo = (const float*)d_in[12]; const float* bo = (const float*)d_in[13];

    const int    Mbig = BB * NN;                 // 32768
    const size_t bigE = (size_t)Mbig * HH;       // elements in [B*N, H]
    const size_t smlE = (size_t)BB * HH;         // elements in [B, H]
    const size_t wE   = (size_t)HH * HH;

    char* ws = (char*)d_ws;
    size_t off = 0;
    auto alloc = [&](size_t bytes) -> char* {
        char* p = ws + off;
        off = (off + bytes + 255) & ~(size_t)255;
        return p;
    };

    bf16* lfb_bf  = (bf16*)alloc(bigE * 2);      // reused as phi after Bfeat
    bf16* Bfeat   = (bf16*)alloc(bigE * 2);
    bf16* g_bf    = (bf16*)alloc(bigE * 2);
    bf16* phi_bf  = lfb_bf;
    bf16* clip_bf = (bf16*)alloc(smlE * 2);
    bf16* Wc_bf   = (bf16*)alloc(wE * 2);
    bf16* Wl_bf   = (bf16*)alloc(wE * 2);
    bf16* Wt_bf   = (bf16*)alloc(wE * 2);
    bf16* Wp_bf   = (bf16*)alloc(wE * 2);
    bf16* Wg_bf   = (bf16*)alloc(wE * 2);
    bf16* Wo_bf   = (bf16*)alloc(wE * 2);
    float* A      = (float*)alloc(smlE * 4);
    bf16* a_bf    = (bf16*)alloc(smlE * 2);
    float* theta  = (float*)alloc(smlE * 4);
    float* sc     = (float*)alloc((size_t)BB * NN * 4);
    float* tbuf   = (float*)alloc(smlE * 4);
    bf16* lr_bf   = (bf16*)alloc(smlE * 2);

    auto cvt = [&](const float* s, bf16* d, size_t n) {
        int ni = (int)n;
        cvt_f32_bf16<<<dim3((unsigned)((n + 255) / 256)), dim3(256), 0, stream>>>(s, d, ni);
    };
    auto gemm = [&](const bf16* X, const bf16* Wm, const float* bias,
                    float* oF, bf16* oB, int M, int addRes) {
        int waves  = (M / 64) * (HH / 64);
        int blocks = (waves + 7) / 8;
        gemm_bf16_wmma<<<dim3(blocks), dim3(256), 0, stream>>>(X, Wm, bias, oF, oB, M, HH, HH, addRes);
    };

    // ---- precision conversions ----
    cvt(lfb,  lfb_bf,  bigE);
    cvt(clip, clip_bf, smlE);
    cvt(Wc, Wc_bf, wE);  cvt(Wl, Wl_bf, wE);  cvt(Wt, Wt_bf, wE);
    cvt(Wp, Wp_bf, wE);  cvt(Wg, Wg_bf, wE);  cvt(Wo, Wo_bf, wE);

    // ---- big GEMMs ----
    gemm(lfb_bf, Wl_bf, bl, nullptr, Bfeat, Mbig, 0);     // Bfeat = lfb @ Wl^T + bl
    gemm(Bfeat,  Wp_bf, bp, nullptr, phi_bf, Mbig, 0);    // phi   (overwrites lfb_bf)
    gemm(Bfeat,  Wg_bf, bg, nullptr, g_bf,  Mbig, 0);     // g
    gemm(clip_bf, Wc_bf, bc, A, nullptr, BB, 0);          // A = clip @ Wc^T + bc

    const float scale = 1.0f / sqrtf((float)HH);
    for (int layer = 0; layer < 3; ++layer) {
        cvt(A, a_bf, smlE);
        gemm(a_bf, Wt_bf, bt, theta, nullptr, BB, 0);     // theta = A @ Wt^T + bt
        {
            int waves = BB * NN;
            scores_kernel<<<dim3((waves * 32 + 255) / 256), dim3(256), 0, stream>>>(theta, phi_bf, sc, scale);
        }
        softmax512<<<dim3(BB), dim3(256), 0, stream>>>(sc);
        attend_kernel<<<dim3((BB * HH + 255) / 256), dim3(256), 0, stream>>>(g_bf, sc, tbuf);
        ln_relu_kernel<<<dim3(BB), dim3(256), 0, stream>>>(tbuf, lr_bf);
        gemm(lr_bf, Wo_bf, bo, A, nullptr, BB, 1);        // A += relu(ln) @ Wo^T + bo
    }

    finalize_kernel<<<dim3((BB * HH + 255) / 256), dim3(256), 0, stream>>>(A, (float*)d_out, BB * HH);
}
